// HybridQLSTM_65481071408683
// MI455X (gfx1250) — matrix operand order repaired and checked by reading, not compiled, not run on hardware
//
#include <hip/hip_runtime.h>
#include <hip/hip_bf16.h>

typedef __attribute__((ext_vector_type(16))) _Float16 v16h;
typedef __attribute__((ext_vector_type(8)))  _Float16 v8h;
typedef __attribute__((ext_vector_type(8)))  float    v8f;

#define TT 2048
#define BB 1024
#define DD 16

static __device__ __forceinline__ v8f wmma_f16(v16h a, v16h b, v8f c) {
  // D = A(16x32 f16) * B(32x16 f16) + C(16x16 f32)
  return __builtin_amdgcn_wmma_f32_16x16x32_f16(
      /*neg_a=*/false, a, /*neg_b=*/false, b,
      /*c_mod=*/(short)0, c, /*reuse_a=*/false, /*reuse_b=*/false);
}

static __device__ __forceinline__ float fast_sigmoid(float x) {
  return 1.0f / (1.0f + __expf(-x));
}
static __device__ __forceinline__ float fast_tanh(float x) {
  // 1 - 2/(e^{2x}+1): stable at both extremes
  return 1.0f - 2.0f / (__expf(2.0f * x) + 1.0f);
}

__global__ __launch_bounds__(128) void qlstm_kernel(
    const float* __restrict__ inputs,
    const float* __restrict__ Wf, const float* __restrict__ bf,
    const float* __restrict__ Wi, const float* __restrict__ bi,
    const float* __restrict__ Wg, const float* __restrict__ bg,
    const float* __restrict__ Wo, const float* __restrict__ bo,
    const float* __restrict__ Q1, const float* __restrict__ qb1,
    const float* __restrict__ Q2, const float* __restrict__ qb2,
    float* __restrict__ out)
{
  // Per-block LDS: h tile (f16 row-major), per-wave transpose scratch, gate exchange
  __shared__ _Float16 sh_h[16 * 16];
  __shared__ _Float16 sh_z[4][16 * 16];
  __shared__ _Float16 sh_q[4][2][16 * 16];
  __shared__ float    sh_gate[4][8 * 32];

  const int tid  = threadIdx.x;
  const int wv   = tid >> 5;   // wave id == gate id: 0=f 1=i 2=g 3=o
  const int lane = tid & 31;
  const int hi   = lane >> 4;  // lane half
  const int nn   = lane & 15;  // N / row-in-tile
  const int row  = nn;
  const int cb   = 8 * hi;     // K sub-block base within 16
  const int tile = blockIdx.x; // batch tile of 16 rows

  const float* Wsel[4] = {Wf, Wi, Wg, Wo};
  const float* bsel[4] = {bf, bi, bg, bo};
  const float* W = Wsel[wv];

  // ---- Resident B fragments (loaded once, f32 -> f16) ----
  // B layout: lane holds column nn, K = 8*hi + e (e < 8), 16 + 8*hi + (e - 8) (e >= 8)
  v16h Bg, Bq1a, Bq1b, Bq2;
  const int kb = 16 * hi;
#pragma unroll
  for (int e = 0; e < 16; ++e) {
    Bg[e]   = (_Float16)W[nn * 32 + cb + e + (e & 8)];    // W.T column nn = W row nn
    Bq1a[e] = (e < 8) ? (_Float16)Q1[nn * 16 + cb + e]    // Q1.T[:,nn]  (K=16, pad e>=8)
                      : (_Float16)0.0f;
    Bq1b[e] = (e < 8) ? (_Float16)Q1[(16 + nn) * 16 + cb + e] // Q1.T[:,16+nn]
                      : (_Float16)0.0f;
    Bq2[e]  = (_Float16)Q2[nn * 32 + cb + e + (e & 8)];   // Q2.T column nn = Q2 row nn
  }
  const float bgate = bsel[wv][nn];
  const float b1lo  = qb1[nn];
  const float b1hi  = qb1[16 + nn];
  const float b2    = qb2[nn];

  // ---- State init: h = 0 (LDS f16), c = 0 (wave0 registers) ----
  sh_h[tid]       = (_Float16)0.0f;   // 128 threads x 2 covers 256 halfs
  sh_h[tid + 128] = (_Float16)0.0f;
  v8f c, hfrag;
#pragma unroll
  for (int j = 0; j < 8; ++j) { c[j] = 0.0f; hfrag[j] = 0.0f; }
  __syncthreads();

  const float* xbase = inputs + ((size_t)tile * 16 + row) * DD + cb;

  for (int t = 0; t < TT; ++t) {
    // ---- Build A = [x | h] fragment ----
    const float* xp = xbase + (size_t)t * BB * DD;
    float4 x0 = *(const float4*)(xp);
    float4 x1 = *(const float4*)(xp + 4);
    if (t + 1 < TT) __builtin_prefetch(xp + BB * DD, 0, 1);
    v8h hrow = *(const v8h*)&sh_h[row * 16 + cb];

    v16h A;
    A[0] = (_Float16)x0.x; A[1] = (_Float16)x0.y;
    A[2] = (_Float16)x0.z; A[3] = (_Float16)x0.w;
    A[4] = (_Float16)x1.x; A[5] = (_Float16)x1.y;
    A[6] = (_Float16)x1.z; A[7] = (_Float16)x1.w;
#pragma unroll
    for (int e = 0; e < 8; ++e) A[8 + e] = hrow[e];

    // ---- z = comb @ W.T + b ----
    v8f z;
#pragma unroll
    for (int j = 0; j < 8; ++j) z[j] = bgate;
    z = wmma_f16(A, Bg, z);

    // D-fragment -> row-major f16 in LDS (transpose bounce)
#pragma unroll
    for (int j = 0; j < 8; ++j)
      sh_z[wv][(j + 8 * hi) * 16 + nn] = (_Float16)z[j];
    asm volatile("s_wait_dscnt 0x0" ::: "memory");

    // ---- q = tanh(z @ Q1.T + qb1), N split into two 16-wide WMMAs (K=16) ----
    v8h za = *(const v8h*)&sh_z[wv][row * 16 + cb];
    v16h A1;
#pragma unroll
    for (int e = 0; e < 8; ++e) { A1[e] = za[e]; A1[8 + e] = (_Float16)0.0f; }

    v8f q0, q1v;
#pragma unroll
    for (int j = 0; j < 8; ++j) { q0[j] = b1lo; q1v[j] = b1hi; }
    q0  = wmma_f16(A1, Bq1a, q0);
    q1v = wmma_f16(A1, Bq1b, q1v);

#pragma unroll
    for (int j = 0; j < 8; ++j) {
      sh_q[wv][0][(j + 8 * hi) * 16 + nn] = (_Float16)fast_tanh(q0[j]);
      sh_q[wv][1][(j + 8 * hi) * 16 + nn] = (_Float16)fast_tanh(q1v[j]);
    }
    asm volatile("s_wait_dscnt 0x0" ::: "memory");

    // ---- ang = q @ Q2.T + qb2 (K=32) ----
    v8h t0 = *(const v8h*)&sh_q[wv][0][row * 16 + cb];
    v8h t1 = *(const v8h*)&sh_q[wv][1][row * 16 + cb];
    v16h A2;
#pragma unroll
    for (int e = 0; e < 8; ++e) { A2[e] = t0[e]; A2[8 + e] = t1[e]; }

    v8f ang;
#pragma unroll
    for (int j = 0; j < 8; ++j) ang[j] = b2;
    ang = wmma_f16(A2, Bq2, ang);

    // ---- qlayer: cumprod(cos) along N (16-lane prefix product), then activation ----
#pragma unroll
    for (int j = 0; j < 8; ++j) {
      float p = __cosf(ang[j]);
#pragma unroll
      for (int off = 1; off < 16; off <<= 1) {
        float u = __shfl_up(p, (unsigned)off, 16);
        if (nn >= off) p *= u;
      }
      p = (wv == 2) ? fast_tanh(p) : fast_sigmoid(p);
      sh_gate[wv][j * 32 + lane] = p;
    }
    __syncthreads();

    // ---- LSTM cell update (wave 0), publish h ----
    if (wv == 0) {
#pragma unroll
      for (int j = 0; j < 8; ++j) {
        float f  = sh_gate[0][j * 32 + lane];
        float i  = sh_gate[1][j * 32 + lane];
        float g  = sh_gate[2][j * 32 + lane];
        float o  = sh_gate[3][j * 32 + lane];
        float cn = f * c[j] + i * g;
        c[j] = cn;
        float h = o * fast_tanh(cn);
        hfrag[j] = h;
        out[((size_t)t * BB + (size_t)tile * 16 + (j + 8 * hi)) * DD + nn] = h;
        sh_h[(j + 8 * hi) * 16 + nn] = (_Float16)h;
      }
    }
    __syncthreads();
  }

  // ---- hx, cx tails ----
  if (wv == 0) {
    const size_t base = (size_t)TT * BB * DD;
#pragma unroll
    for (int j = 0; j < 8; ++j) {
      size_t idx = ((size_t)tile * 16 + (j + 8 * hi)) * DD + nn;
      out[base + idx] = hfrag[j];
      out[base + (size_t)BB * DD + idx] = c[j];
    }
  }
}

extern "C" void kernel_launch(void* const* d_in, const int* in_sizes, int n_in,
                              void* d_out, int out_size, void* d_ws, size_t ws_size,
                              hipStream_t stream) {
  (void)in_sizes; (void)n_in; (void)out_size; (void)d_ws; (void)ws_size;
  const float* inputs = (const float*)d_in[0];
  const float* Wf  = (const float*)d_in[1];
  const float* bf  = (const float*)d_in[2];
  const float* Wi  = (const float*)d_in[3];
  const float* bi  = (const float*)d_in[4];
  const float* Wg  = (const float*)d_in[5];
  const float* bg  = (const float*)d_in[6];
  const float* Wo  = (const float*)d_in[7];
  const float* bo  = (const float*)d_in[8];
  const float* Q1  = (const float*)d_in[9];
  const float* qb1 = (const float*)d_in[10];
  const float* Q2  = (const float*)d_in[11];
  const float* qb2 = (const float*)d_in[12];
  float* out = (float*)d_out;

  dim3 grid(BB / 16);   // 64 batch tiles
  dim3 block(128);      // 4 wave32 waves: one per gate
  qlstm_kernel<<<grid, block, 0, stream>>>(inputs, Wf, bf, Wi, bi, Wg, bg,
                                           Wo, bo, Q1, qb1, Q2, qb2, out);
}
